// KroneckerDecompAttention_45457933861377
// MI455X (gfx1250) — matrix operand order repaired and checked by reading, not compiled, not run on hardware
//
#include <hip/hip_runtime.h>
#include <hip/hip_bf16.h>
#include <math.h>

// ---------------- problem constants (from reference setup_inputs) ----------------
#define BH     32          // b*h = 2*16
#define DHEAD  128
#define NGRP   8           // n_query_groups == n_key_groups
#define PLEN   1024        // 8192 / 8
#define NFULL  8192
#define MSAMP  34          // max(1, int(1024/30))
#define NSUB   (NGRP*MSAMP)   // 272 (divisible by 16 -> 17 query tiles)
#define EPSF   1.1920929e-07f

typedef __attribute__((ext_vector_type(16))) _Float16 v16h;
typedef __attribute__((ext_vector_type(8)))  float    v8f;

// ---------------- half-wave (16-lane) reductions, wave32 ----------------
__device__ inline float half_max16(float v) {
  v = fmaxf(v, __shfl_xor(v, 1, 32));
  v = fmaxf(v, __shfl_xor(v, 2, 32));
  v = fmaxf(v, __shfl_xor(v, 4, 32));
  v = fmaxf(v, __shfl_xor(v, 8, 32));
  return v;
}
__device__ inline float half_sum16(float v) {
  v += __shfl_xor(v, 1, 32);
  v += __shfl_xor(v, 2, 32);
  v += __shfl_xor(v, 4, 32);
  v += __shfl_xor(v, 8, 32);
  return v;
}

// ---------------- group means: q_rep, k_rep, v_mean ----------------
__global__ void __launch_bounds__(256)
means_kernel(const float* __restrict__ q, const float* __restrict__ k,
             const float* __restrict__ v, float* __restrict__ q_rep,
             float* __restrict__ k_rep, float* __restrict__ v_mean) {
  size_t idx = (size_t)blockIdx.x * 256 + threadIdx.x;
  size_t total = (size_t)BH * PLEN * DHEAD;
  if (idx >= total) return;
  int dd = (int)(idx % DHEAD);
  int i  = (int)((idx / DHEAD) % PLEN);
  int bh = (int)(idx / ((size_t)PLEN * DHEAD));
  size_t base = (size_t)bh * NFULL * DHEAD + (size_t)i * DHEAD + dd;
  float sq = 0.f, sk = 0.f, sv = 0.f;
#pragma unroll
  for (int g = 0; g < NGRP; ++g) {
    size_t o = base + (size_t)g * PLEN * DHEAD;
    sq += q[o]; sk += k[o]; sv += v[o];
  }
  const float inv = 1.0f / NGRP;
  q_rep[idx] = sq * inv; k_rep[idx] = sk * inv; v_mean[idx] = sv * inv;
}

// ---------------- deterministic Gumbel top-k over residual row norms ----------------
__device__ inline unsigned long long splitmix64(unsigned long long x) {
  x += 0x9E3779B97F4A7C15ull;
  x = (x ^ (x >> 30)) * 0xBF58476D1CE4E5B9ull;
  x = (x ^ (x >> 27)) * 0x94D049BB133111EBull;
  return x ^ (x >> 31);
}

__global__ void __launch_bounds__(256)
topk_kernel(const float* __restrict__ data, const float* __restrict__ rep,
            int* __restrict__ idx_out, unsigned seed) {
  __shared__ float sc[PLEN];
  __shared__ float rv[256];
  __shared__ int   ri[256];
  const int bh = blockIdx.x / NGRP, g = blockIdx.x % NGRP;
  const int tid = threadIdx.x;
  const float* drows = data + ((size_t)bh * NFULL + (size_t)g * PLEN) * DHEAD;
  const float* rrows = rep + (size_t)bh * PLEN * DHEAD;
  for (int j = tid; j < PLEN; j += 256) {
    float ss = 0.f;
    const float* a = drows + (size_t)j * DHEAD;
    const float* b = rrows + (size_t)j * DHEAD;
    for (int dd = 0; dd < DHEAD; ++dd) { float df = a[dd] - b[dd]; ss += df * df; }
    float prob = sqrtf(ss) * 100.f + EPSF;
    unsigned long long h = splitmix64(((unsigned long long)seed << 32) ^
                                      ((unsigned long long)blockIdx.x << 16) ^ (unsigned)j);
    float u = ((float)(h >> 40) + 0.5f) * (1.0f / 16777216.0f);
    float gum = -logf(-logf(u));
    sc[j] = logf(prob) + gum;
  }
  __syncthreads();
  int* outp = idx_out + (size_t)blockIdx.x * MSAMP;
  for (int s = 0; s < MSAMP; ++s) {
    float bv = -1e38f; int bi = 0;
    for (int j = tid; j < PLEN; j += 256)
      if (sc[j] > bv) { bv = sc[j]; bi = j; }
    rv[tid] = bv; ri[tid] = bi;
    __syncthreads();
    for (int w = 128; w > 0; w >>= 1) {
      if (tid < w && rv[tid + w] > rv[tid]) { rv[tid] = rv[tid + w]; ri[tid] = ri[tid + w]; }
      __syncthreads();
    }
    if (tid == 0) { outp[s] = ri[0]; sc[ri[0]] = -1e38f; }
    __syncthreads();
  }
}

// ---------------- gathers ----------------
__global__ void __launch_bounds__(256)
gather_k_kernel(const float* __restrict__ key, const float* __restrict__ val,
                const float* __restrict__ k_rep, const int* __restrict__ k_idx,
                float* __restrict__ k_sub, float* __restrict__ v_sub,
                float* __restrict__ krep_sub) {
  size_t idx = (size_t)blockIdx.x * 256 + threadIdx.x;
  size_t total = (size_t)BH * NSUB * DHEAD;
  if (idx >= total) return;
  int dd = (int)(idx % DHEAD);
  int t  = (int)((idx / DHEAD) % NSUB);
  int bh = (int)(idx / ((size_t)NSUB * DHEAD));
  int g = t / MSAMP, s = t % MSAMP;
  int j = k_idx[(bh * NGRP + g) * MSAMP + s];
  size_t src = ((size_t)bh * NFULL + (size_t)g * PLEN + j) * DHEAD + dd;
  k_sub[idx] = key[src];
  v_sub[idx] = val[src];
  krep_sub[idx] = k_rep[((size_t)bh * PLEN + j) * DHEAD + dd];
}

__global__ void __launch_bounds__(256)
gather_q_kernel(const float* __restrict__ query, const int* __restrict__ q_idx,
                float* __restrict__ q_sub) {
  size_t idx = (size_t)blockIdx.x * 256 + threadIdx.x;
  size_t total = (size_t)BH * NSUB * DHEAD;
  if (idx >= total) return;
  int dd = (int)(idx % DHEAD);
  int t  = (int)((idx / DHEAD) % NSUB);
  int bh = (int)(idx / ((size_t)NSUB * DHEAD));
  int g = t / MSAMP, s = t % MSAMP;
  int j = q_idx[(bh * NGRP + g) * MSAMP + s];
  q_sub[idx] = query[((size_t)bh * NFULL + (size_t)g * PLEN + j) * DHEAD + dd];
}

// ---------------- generic flash attention, one wave = one 16-query tile ----------------
// Layouts per CDNA5 ISA 7.12.2 (wave32). d=128 resident in 8 f32 16x16 accumulators.
__global__ void __launch_bounds__(32)
fattn_kernel(const float* __restrict__ Q, const float* __restrict__ K,
             const float* __restrict__ V, float* __restrict__ O,
             float* __restrict__ LSE, int nq, int nk, float scale, float lse_add) {
  __shared__ _Float16 lp[16 * 32];   // P relayout buffer (C-layout -> A-layout)
  const int bh   = blockIdx.y;
  const int qt   = blockIdx.x;
  const int lane = threadIdx.x;
  const int half = lane >> 4;        // 0 or 1
  const int ln   = lane & 15;

  Q   += (size_t)bh * nq * DHEAD;
  K   += (size_t)bh * nk * DHEAD;
  V   += (size_t)bh * nk * DHEAD;
  O   += (size_t)bh * nq * DHEAD;
  LSE += (size_t)bh * nq;

  // Q tile -> A-layout f16 (4 chunks of K=32 features), held in registers for all key blocks
  v16h qa[4];
  {
    const float* qrow = Q + (size_t)(qt * 16 + ln) * DHEAD;
#pragma unroll
    for (int c = 0; c < 4; ++c) {
      const int f0 = c * 32 + half * 16;
      v16h a;
#pragma unroll
      for (int e = 0; e < 8; ++e) a[e]     = (_Float16)qrow[f0 + e];
#pragma unroll
      for (int e = 0; e < 8; ++e) a[e + 8] = (_Float16)qrow[f0 + 8 + e];
      qa[c] = a;
    }
  }

  float mrow[8], lrow[8];
  v8f acc[8];
#pragma unroll
  for (int r = 0; r < 8; ++r) { mrow[r] = -1e30f; lrow[r] = 0.f; }
#pragma unroll
  for (int nt = 0; nt < 8; ++nt) acc[nt] = (v8f){};

  const int nkb = (nk + 31) / 32;
  for (int kb = 0; kb < nkb; ++kb) {
    const int base = kb * 32;
    // ---- S = Q K^T for 32 keys as two 16x16 tiles
    v8f s0 = (v8f){}, s1 = (v8f){};
    int kr0 = base + ln;      if (kr0 >= nk) kr0 = nk - 1;
    int kr1 = base + 16 + ln; if (kr1 >= nk) kr1 = nk - 1;
    const float* krow0 = K + (size_t)kr0 * DHEAD;
    const float* krow1 = K + (size_t)kr1 * DHEAD;
#pragma unroll
    for (int c = 0; c < 4; ++c) {
      const int f0 = c * 32 + half * 16;
      v16h b0, b1;
#pragma unroll
      for (int e = 0; e < 16; ++e) b0[e] = (_Float16)krow0[f0 + e];
#pragma unroll
      for (int e = 0; e < 16; ++e) b1[e] = (_Float16)krow1[f0 + e];
      s0 = __builtin_amdgcn_wmma_f32_16x16x32_f16(false, qa[c], false, b0, (short)0, s0, false, false);
      s1 = __builtin_amdgcn_wmma_f32_16x16x32_f16(false, qa[c], false, b1, (short)0, s1, false, false);
    }
    const bool ok0 = (base + ln) < nk;
    const bool ok1 = (base + 16 + ln) < nk;
    // ---- online softmax (row state replicated across 16-lane half)
    float p0[8], p1[8], alpha[8];
#pragma unroll
    for (int r = 0; r < 8; ++r) {
      float v0 = ok0 ? s0[r] * scale : -1e30f;
      float v1 = ok1 ? s1[r] * scale : -1e30f;
      float t  = half_max16(fmaxf(v0, v1));
      float mn = fmaxf(mrow[r], t);
      alpha[r] = __expf(mrow[r] - mn);
      p0[r] = __expf(v0 - mn);
      p1[r] = __expf(v1 - mn);
      float rs = half_sum16(p0[r] + p1[r]);
      lrow[r] = lrow[r] * alpha[r] + rs;
      mrow[r] = mn;
    }
#pragma unroll
    for (int nt = 0; nt < 8; ++nt)
#pragma unroll
      for (int r = 0; r < 8; ++r) acc[nt][r] *= alpha[r];
    // ---- P: C-layout f32 -> LDS row-major 16x32 f16 -> A-layout
    __syncthreads();   // single-wave block: degenerates to NOP-barrier
#pragma unroll
    for (int r = 0; r < 8; ++r) {
      const int m = r + 8 * half;
      lp[m * 32 + ln]      = (_Float16)p0[r];
      lp[m * 32 + 16 + ln] = (_Float16)p1[r];
    }
    __syncthreads();
    v16h pa;
#pragma unroll
    for (int e = 0; e < 8; ++e) pa[e]     = lp[ln * 32 + half * 16 + e];
#pragma unroll
    for (int e = 0; e < 8; ++e) pa[e + 8] = lp[ln * 32 + half * 16 + 8 + e];
    // ---- O += P(16x32) @ V(32x128), 8 output feature tiles
#pragma unroll
    for (int nt = 0; nt < 8; ++nt) {
      v16h vb;
#pragma unroll
      for (int e = 0; e < 16; ++e) {
        int kr = base + half * 16 + e; if (kr >= nk) kr = nk - 1;
        vb[e] = (_Float16)V[(size_t)kr * DHEAD + nt * 16 + ln];
      }
      acc[nt] = __builtin_amdgcn_wmma_f32_16x16x32_f16(false, pa, false, vb, (short)0, acc[nt], false, false);
    }
  }
  // ---- finalize: O /= rowsum; LSE = m + log(l) + lse_add
#pragma unroll
  for (int r = 0; r < 8; ++r) {
    const int m = r + 8 * half;
    const float invl = 1.0f / lrow[r];
#pragma unroll
    for (int nt = 0; nt < 8; ++nt)
      O[(size_t)(qt * 16 + m) * DHEAD + nt * 16 + ln] = acc[nt][r] * invl;
    if (ln == 0)
      LSE[qt * 16 + m] = mrow[r] + logf(lrow[r]) + lse_add;
  }
}

// ---------------- log-space combine (subtract then add), split attn/lse passes ----------------
__global__ void __launch_bounds__(256)
combine_sub_attn_kernel(float* __restrict__ attn0, const float* __restrict__ lse0,
                        const float* __restrict__ adel, const float* __restrict__ ldel) {
  size_t idx = (size_t)blockIdx.x * 256 + threadIdx.x;
  if (idx >= (size_t)BH * PLEN * DHEAD) return;
  size_t row = idx / DHEAD;
  float c = __expf(ldel[row] - lse0[row]);
  attn0[idx] = (attn0[idx] - c * adel[idx]) / (1.0f - c);
}
__global__ void __launch_bounds__(256)
combine_sub_lse_kernel(float* __restrict__ lse0, const float* __restrict__ ldel) {
  size_t row = (size_t)blockIdx.x * 256 + threadIdx.x;
  if (row >= (size_t)BH * PLEN) return;
  float c = __expf(ldel[row] - lse0[row]);
  lse0[row] = lse0[row] + log1pf(-c);
}
__global__ void __launch_bounds__(256)
combine_add_attn_kernel(float* __restrict__ attn0, const float* __restrict__ lse0,
                        const float* __restrict__ aadd, const float* __restrict__ ladd) {
  size_t idx = (size_t)blockIdx.x * 256 + threadIdx.x;
  if (idx >= (size_t)BH * PLEN * DHEAD) return;
  size_t row = idx / DHEAD;
  float c = 1.0f / (1.0f + __expf(ladd[row] - lse0[row]));  // sigmoid(lse0 - ladd)
  attn0[idx] = c * attn0[idx] + (1.0f - c) * aadd[idx];
}
__global__ void __launch_bounds__(256)
combine_add_lse_kernel(float* __restrict__ lse0, const float* __restrict__ ladd) {
  size_t row = (size_t)blockIdx.x * 256 + threadIdx.x;
  if (row >= (size_t)BH * PLEN) return;
  float a = lse0[row], b = ladd[row];
  float mx = fmaxf(a, b);
  lse0[row] = mx + log1pf(__expf(fminf(a, b) - mx));
}

// ---------------- output broadcast + sampled-row scatter ----------------
__global__ void __launch_bounds__(256)
broadcast_kernel(const float* __restrict__ attn1, const float* __restrict__ lse1,
                 float* __restrict__ out_attn, float* __restrict__ out_lse) {
  size_t idx = (size_t)blockIdx.x * 256 + threadIdx.x;
  if (idx >= (size_t)BH * NFULL * DHEAD) return;
  int dd = (int)(idx % DHEAD);
  size_t rowfull = (idx / DHEAD) % NFULL;
  int bh = (int)(idx / ((size_t)NFULL * DHEAD));
  int i = (int)(rowfull % PLEN);
  out_attn[idx] = attn1[((size_t)bh * PLEN + i) * DHEAD + dd];
  if (dd == 0) out_lse[(size_t)bh * NFULL + rowfull] = lse1[(size_t)bh * PLEN + i];
}

__global__ void __launch_bounds__(256)
scatter_kernel(const float* __restrict__ a2, const float* __restrict__ l2,
               const int* __restrict__ q_idx,
               float* __restrict__ out_attn, float* __restrict__ out_lse) {
  size_t idx = (size_t)blockIdx.x * 256 + threadIdx.x;
  if (idx >= (size_t)BH * NSUB * DHEAD) return;
  int dd = (int)(idx % DHEAD);
  int t  = (int)((idx / DHEAD) % NSUB);
  int bh = (int)(idx / ((size_t)NSUB * DHEAD));
  int g = t / MSAMP, s = t % MSAMP;
  int j = q_idx[(bh * NGRP + g) * MSAMP + s];
  size_t dstrow = (size_t)bh * NFULL + (size_t)g * PLEN + j;
  out_attn[dstrow * DHEAD + dd] = a2[idx];
  if (dd == 0) out_lse[dstrow] = l2[(size_t)bh * NSUB + t];
}

// ---------------- launch ----------------
extern "C" void kernel_launch(void* const* d_in, const int* in_sizes, int n_in,
                              void* d_out, int out_size, void* d_ws, size_t ws_size,
                              hipStream_t stream) {
  const float* q = (const float*)d_in[0];
  const float* k = (const float*)d_in[1];
  const float* v = (const float*)d_in[2];
  float* out_attn = (float*)d_out;
  float* out_lse  = out_attn + (size_t)BH * NFULL * DHEAD;

  float* ws = (float*)d_ws;
  size_t off = 0;
  float* q_rep   = ws + off; off += (size_t)BH * PLEN * DHEAD;
  float* k_rep   = ws + off; off += (size_t)BH * PLEN * DHEAD;
  float* v_mean  = ws + off; off += (size_t)BH * PLEN * DHEAD;
  float* attn0   = ws + off; off += (size_t)BH * PLEN * DHEAD;   // attn_p0 -> attn_p1 (in place)
  float* a_tmp   = ws + off; off += (size_t)BH * PLEN * DHEAD;   // a_del then a_add
  float* lse0    = ws + off; off += (size_t)BH * PLEN;
  float* l_tmp   = ws + off; off += (size_t)BH * PLEN;
  float* k_sub   = ws + off; off += (size_t)BH * NSUB * DHEAD;
  float* v_sub   = ws + off; off += (size_t)BH * NSUB * DHEAD;
  float* krepsub = ws + off; off += (size_t)BH * NSUB * DHEAD;
  float* q_sub   = ws + off; off += (size_t)BH * NSUB * DHEAD;
  float* a2buf   = ws + off; off += (size_t)BH * NSUB * DHEAD;
  float* l2buf   = ws + off; off += (size_t)BH * NSUB;
  int*   k_idx   = (int*)(ws + off); off += (size_t)BH * NGRP * MSAMP;
  int*   q_idx   = (int*)(ws + off); off += (size_t)BH * NGRP * MSAMP;

  const float scale   = 0.08838834764831845f;  // 128^-0.5
  const float log_ng  = 2.0794415416798357f;   // log(8)

  const int blkRep  = (int)(((size_t)BH * PLEN * DHEAD) / 256);    // 16384
  const int blkFull = (int)(((size_t)BH * NFULL * DHEAD) / 256);   // 131072
  const int blkSub  = (int)(((size_t)BH * NSUB * DHEAD) / 256);    // 4352
  const int blkLse  = (int)(((size_t)BH * PLEN + 255) / 256);      // 128

  // 1) group means
  means_kernel<<<blkRep, 256, 0, stream>>>(q, k, v, q_rep, k_rep, v_mean);
  // 2) key sampling + gathers
  topk_kernel<<<BH * NGRP, 256, 0, stream>>>(k, k_rep, k_idx, 0x1234ABCDu);
  gather_k_kernel<<<blkSub, 256, 0, stream>>>(k, v, k_rep, k_idx, k_sub, v_sub, krepsub);
  // 3) p0: representative attention; lse += log(n_g), V already mean (== /n_g)
  fattn_kernel<<<dim3(PLEN / 16, BH), 32, 0, stream>>>(q_rep, k_rep, v_mean, attn0, lse0,
                                                       PLEN, PLEN, scale, log_ng);
  // 4) p1 subtract: a_del = attn(q_rep, k_rep_sub, v_sub)
  fattn_kernel<<<dim3(PLEN / 16, BH), 32, 0, stream>>>(q_rep, krepsub, v_sub, a_tmp, l_tmp,
                                                       PLEN, NSUB, scale, 0.0f);
  combine_sub_attn_kernel<<<blkRep, 256, 0, stream>>>(attn0, lse0, a_tmp, l_tmp);
  combine_sub_lse_kernel<<<blkLse, 256, 0, stream>>>(lse0, l_tmp);
  // 5) p1 add: a_add = attn(q_rep, k_sub, v_sub)
  fattn_kernel<<<dim3(PLEN / 16, BH), 32, 0, stream>>>(q_rep, k_sub, v_sub, a_tmp, l_tmp,
                                                       PLEN, NSUB, scale, 0.0f);
  combine_add_attn_kernel<<<blkRep, 256, 0, stream>>>(attn0, lse0, a_tmp, l_tmp);
  combine_add_lse_kernel<<<blkLse, 256, 0, stream>>>(lse0, l_tmp);
  // 6) p2: query sampling, gather, exact attention vs full K/V
  topk_kernel<<<BH * NGRP, 256, 0, stream>>>(q, q_rep, q_idx, 0x5678EF01u);
  gather_q_kernel<<<blkSub, 256, 0, stream>>>(q, q_idx, q_sub);
  fattn_kernel<<<dim3(NSUB / 16, BH), 32, 0, stream>>>(q_sub, k, v, a2buf, l2buf,
                                                       NSUB, NFULL, scale, 0.0f);
  // 7) assemble output: broadcast p1 result to all groups, then scatter exact rows
  broadcast_kernel<<<blkFull, 256, 0, stream>>>(attn0, lse0, out_attn, out_lse);
  scatter_kernel<<<blkSub, 256, 0, stream>>>(a2buf, l2buf, q_idx, out_attn, out_lse);
}
